// MemoryOperation_80161269613278
// MI455X (gfx1250) — matrix operand order repaired and checked by reading, not compiled, not run on hardware
//
#include <hip/hip_runtime.h>
#include <hip/hip_bf16.h>

#define DIM   128
#define EF    172
#define TD    100
#define MSG   528          // 2*DIM + EF + TD
#define MSGP  544          // padded to 17*32 for K-tiling
#define GATE  384          // 3*DIM

typedef __bf16 v16bf __attribute__((ext_vector_type(16)));
typedef __bf16 v8bf  __attribute__((ext_vector_type(8)));
typedef float  v8f   __attribute__((ext_vector_type(8)));

union AFrag { v16bf v; v8bf h[2]; };

__device__ __forceinline__ unsigned short f2bf(float f) {
    unsigned int u = __float_as_uint(f);
    unsigned int r = u + 0x7FFFu + ((u >> 16) & 1u);   // round-to-nearest-even
    return (unsigned short)(r >> 16);
}

// ---------------------------------------------------------------------------
// k_prep: full-table copy, ts copy, zero argmax keys, weights -> bf16 (padded)
// ---------------------------------------------------------------------------
__global__ void k_prep(const float* __restrict__ mem, const float* __restrict__ lt,
                       const float* __restrict__ wih_f, const float* __restrict__ whh_f,
                       float* __restrict__ out_mem, float* __restrict__ out_ts,
                       unsigned long long* __restrict__ keys,
                       unsigned short* __restrict__ wih_b, unsigned short* __restrict__ whh_b,
                       long long n_mem4, int n_total, int n_g)
{
    long long tid = (long long)blockIdx.x * blockDim.x + threadIdx.x;
    long long stride = (long long)gridDim.x * blockDim.x;

    const float4* s4 = (const float4*)mem;
    float4* d4 = (float4*)out_mem;
    for (long long i = tid; i < n_mem4; i += stride) d4[i] = s4[i];
    for (long long i = tid; i < n_total; i += stride) out_ts[i] = lt[i];
    for (long long i = tid; i < n_g; i += stride) keys[i] = 0ull;
    for (long long i = tid; i < (long long)GATE * MSGP; i += stride) {
        int j = (int)(i / MSGP), k = (int)(i % MSGP);
        wih_b[i] = f2bf(k < MSG ? wih_f[(long long)j * MSG + k] : 0.0f);
    }
    for (long long i = tid; i < (long long)GATE * DIM; i += stride)
        whh_b[i] = f2bf(whh_f[i]);
}

// ---------------------------------------------------------------------------
// k_edges: deterministic segment-argmax via packed (ts, ~edge) u64 atomicMax
// ---------------------------------------------------------------------------
__global__ void k_edges(const float* __restrict__ edge_ts, const int* __restrict__ dst,
                        unsigned long long* __restrict__ keys, int n_e)
{
    int e = blockIdx.x * blockDim.x + threadIdx.x;
    if (e < n_e) {
        unsigned long long key =
            ((unsigned long long)__float_as_uint(edge_ts[e]) << 32) |
            (unsigned long long)(0xFFFFFFFFu - (unsigned int)e);
        atomicMax(keys + dst[e], key);
    }
}

// ---------------------------------------------------------------------------
// k_fused: per 16-node tile -> build bf16 msg/h in LDS -> WMMA GEMMs -> GRU
// ---------------------------------------------------------------------------
__global__ void __launch_bounds__(256)
k_fused(const float* __restrict__ memory, const float* __restrict__ last_t,
        const float* __restrict__ edge_feats, const float* __restrict__ edge_ts,
        const float* __restrict__ time_w, const float* __restrict__ time_b,
        const float* __restrict__ b_ih, const float* __restrict__ b_hh,
        const int* __restrict__ nid, const int* __restrict__ src, const int* __restrict__ dst,
        const unsigned long long* __restrict__ keys,
        const unsigned short* __restrict__ wih, const unsigned short* __restrict__ whh,
        float* __restrict__ out_mem, float* __restrict__ out_ts, int n_g)
{
    // Phase A storage (bf16 tiles) is reused as Phase C storage (f32 gates).
    __shared__ alignas(32) unsigned char smem_raw[2 * 16 * GATE * 4];   // 49152 B
    unsigned short* s_msg = (unsigned short*)smem_raw;                  // 16 x MSGP bf16
    unsigned short* s_hm  = (unsigned short*)(smem_raw + 16 * MSGP * 2);// 16 x DIM  bf16
    float* s_gi = (float*)smem_raw;                                     // 16 x GATE f32
    float* s_gh = (float*)(smem_raw + 16 * GATE * 4);                   // 16 x GATE f32

    __shared__ int   s_has[16], s_best[16], s_srcg[16], s_dstg[16], s_gid[16];
    __shared__ float s_delta[16];

    const int t  = threadIdx.x;
    const int n0 = blockIdx.x * 16;

    // ---- decode best edge per node ----
    if (t < 16) {
        int n = n0 + t;
        int has = 0, best = 0, sg = 0, dg = 0, gid = 0;
        float delta = 0.0f;
        if (n < n_g) {
            gid = nid[n];
            unsigned long long key = keys[n];
            has = (key != 0ull);
            best = has ? (int)(0xFFFFFFFFu - (unsigned int)(key & 0xFFFFFFFFull)) : 0;
            float mts = __uint_as_float((unsigned int)(key >> 32));
            sg = nid[src[best]];
            dg = nid[dst[best]];
            delta = edge_ts[best] - last_t[sg];
            out_ts[gid] = has ? mts : last_t[gid];
        }
        s_has[t] = has; s_best[t] = best; s_srcg[t] = sg; s_dstg[t] = dg;
        s_gid[t] = gid; s_delta[t] = delta;
    }
    __syncthreads();

    // ---- build message tile (bf16) ----
    for (int idx = t; idx < 16 * MSGP; idx += 256) {
        int m = idx / MSGP, c = idx % MSGP;
        float v = 0.0f;
        if ((n0 + m) < n_g && s_has[m]) {
            if (c < DIM)              v = memory[(long long)s_srcg[m] * DIM + c];
            else if (c < 2 * DIM)     v = memory[(long long)s_dstg[m] * DIM + (c - DIM)];
            else if (c < 2 * DIM + EF) v = edge_feats[(long long)s_best[m] * EF + (c - 2 * DIM)];
            else if (c < MSG) {
                int k = c - (2 * DIM + EF);
                v = cosf(s_delta[m] * time_w[k] + time_b[k]);
            }
        }
        s_msg[idx] = f2bf(v);
    }
    // ---- build h tile (bf16) ----
    for (int idx = t; idx < 16 * DIM; idx += 256) {
        int m = idx / DIM, c = idx % DIM;
        float v = ((n0 + m) < n_g) ? memory[(long long)s_gid[m] * DIM + c] : 0.0f;
        s_hm[idx] = f2bf(v);
    }
    __syncthreads();

    // ---- WMMA GEMMs: gi = msg @ W_ih^T   gh = h @ W_hh^T ----
    const int wave = t >> 5;         // 0..7, each owns 3 column tiles (j0 = (wave*3+i)*16)
    const int lane = t & 31;
    const int half = lane >> 4;      // K-half selector for A/B lane layouts
    const int l16  = lane & 15;

    v8f acc_gi[3] = {}, acc_gh[3] = {};

    for (int kt = 0; kt < MSGP / 32; ++kt) {
        int kbase = kt * 32 + half * 16;
        AFrag a;
        a.h[0] = *(const v8bf*)(s_msg + l16 * MSGP + kbase);
        a.h[1] = *(const v8bf*)(s_msg + l16 * MSGP + kbase + 8);
#pragma unroll
        for (int i = 0; i < 3; ++i) {
            int j = (wave * 3 + i) * 16 + l16;
            v16bf bm = *(const v16bf*)(wih + (long long)j * MSGP + kt * 32 + half * 16);
            acc_gi[i] = __builtin_amdgcn_wmma_f32_16x16x32_bf16(
                false, a.v, false, bm, (short)0, acc_gi[i], false, false);
        }
    }
    for (int kt = 0; kt < DIM / 32; ++kt) {
        int kbase = kt * 32 + half * 16;
        AFrag a;
        a.h[0] = *(const v8bf*)(s_hm + l16 * DIM + kbase);
        a.h[1] = *(const v8bf*)(s_hm + l16 * DIM + kbase + 8);
#pragma unroll
        for (int i = 0; i < 3; ++i) {
            int j = (wave * 3 + i) * 16 + l16;
            v16bf bm = *(const v16bf*)(whh + (long long)j * DIM + kt * 32 + half * 16);
            acc_gh[i] = __builtin_amdgcn_wmma_f32_16x16x32_bf16(
                false, a.v, false, bm, (short)0, acc_gh[i], false, false);
        }
    }

    __syncthreads();   // done reading bf16 tiles; reuse LDS for f32 gates
#pragma unroll
    for (int i = 0; i < 3; ++i) {
        int j0 = (wave * 3 + i) * 16;
#pragma unroll
        for (int r = 0; r < 8; ++r) {
            int m = r + 8 * half;     // C/D layout: lanes 0-15 rows 0-7, lanes 16-31 rows 8-15
            s_gi[m * GATE + j0 + l16] = acc_gi[i][r];
            s_gh[m * GATE + j0 + l16] = acc_gh[i][r];
        }
    }
    __syncthreads();

    // ---- GRU epilogue (f32), scatter to output ----
    {
        int m = t >> 4;
        int dl = t & 15;
        int n = n0 + m;
        if (n < n_g) {
            int has = s_has[m];
            int gid = s_gid[m];
#pragma unroll
            for (int q = 0; q < 8; ++q) {
                int d = dl + 16 * q;
                float h_old = memory[(long long)gid * DIM + d];
                float outv = h_old;
                if (has) {
                    float ir = s_gi[m * GATE + d]           + b_ih[d];
                    float iz = s_gi[m * GATE + d + DIM]     + b_ih[d + DIM];
                    float in_ = s_gi[m * GATE + d + 2 * DIM] + b_ih[d + 2 * DIM];
                    float hr = s_gh[m * GATE + d]           + b_hh[d];
                    float hz = s_gh[m * GATE + d + DIM]     + b_hh[d + DIM];
                    float hn = s_gh[m * GATE + d + 2 * DIM] + b_hh[d + 2 * DIM];
                    float r = 1.0f / (1.0f + expf(-(ir + hr)));
                    float z = 1.0f / (1.0f + expf(-(iz + hz)));
                    float nn = tanhf(in_ + r * hn);
                    outv = (1.0f - z) * nn + z * h_old;
                }
                out_mem[(long long)gid * DIM + d] = outv;
            }
        }
    }
}

// ---------------------------------------------------------------------------
extern "C" void kernel_launch(void* const* d_in, const int* in_sizes, int n_in,
                              void* d_out, int out_size, void* d_ws, size_t ws_size,
                              hipStream_t stream)
{
    const float* memory     = (const float*)d_in[0];
    const float* last_t     = (const float*)d_in[1];
    const float* edge_feats = (const float*)d_in[2];
    const float* edge_ts    = (const float*)d_in[3];
    const float* time_w     = (const float*)d_in[4];
    const float* time_b     = (const float*)d_in[5];
    const float* W_ih       = (const float*)d_in[6];
    const float* W_hh       = (const float*)d_in[7];
    const float* b_ih       = (const float*)d_in[8];
    const float* b_hh       = (const float*)d_in[9];
    const int*   nid        = (const int*)d_in[10];
    const int*   src        = (const int*)d_in[11];
    const int*   dst        = (const int*)d_in[12];

    const int n_total = in_sizes[0] / DIM;
    const int n_g     = in_sizes[10];
    const int n_e     = in_sizes[3];

    float* out_mem = (float*)d_out;
    float* out_ts  = out_mem + (long long)n_total * DIM;

    unsigned char* ws = (unsigned char*)d_ws;
    unsigned long long* keys = (unsigned long long*)ws;
    size_t off = ((size_t)n_g * 8 + 63) & ~(size_t)63;
    unsigned short* wih_b = (unsigned short*)(ws + off);
    off += (size_t)GATE * MSGP * 2;             // 417792 B (64-aligned)
    unsigned short* whh_b = (unsigned short*)(ws + off);

    long long n_mem4 = (long long)n_total * DIM / 4;

    k_prep<<<4096, 256, 0, stream>>>(memory, last_t, W_ih, W_hh,
                                     out_mem, out_ts, keys, wih_b, whh_b,
                                     n_mem4, n_total, n_g);
    k_edges<<<(n_e + 255) / 256, 256, 0, stream>>>(edge_ts, dst, keys, n_e);
    k_fused<<<(n_g + 15) / 16, 256, 0, stream>>>(memory, last_t, edge_feats, edge_ts,
                                                 time_w, time_b, b_ih, b_hh,
                                                 nid, src, dst, keys, wih_b, whh_b,
                                                 out_mem, out_ts, n_g);
}
